// MLPAttention_69793218560318
// MI455X (gfx1250) — matrix operand order repaired and checked by reading, not compiled, not run on hardware
//
#include <hip/hip_runtime.h>
#include <hip/hip_bf16.h>

// ---------------------------------------------------------------------------
// MLP attention for MI455X (gfx1250, wave32, WMMA).
//  hidden = tanh(ctx@Wc + q@Wq + b1); score = hidden@w2 + b2
//  p = softmax(mask ? score : -1e4); out = p@ctx
// Dominant GEMM (68.7 GFLOP) on v_wmma_f32_16x16x32_bf16, N-blocked by 4:
// per kt one A fragment feeds 4 WMMAs into 4 accumulator chains
// (DS traffic 40B/WMMA instead of 64B, batched dscnt waits).
// B (packed Wc) staged as contiguous 64KB half-group chunks into LDS,
// double-buffered via global_load_async_to_lds_b128 (ASYNCcnt).
// ---------------------------------------------------------------------------

typedef __attribute__((ext_vector_type(16))) __bf16 bf16x16;
typedef __attribute__((ext_vector_type(8)))  __bf16 bf16x8;
typedef __attribute__((ext_vector_type(8)))  float  f32x8;
typedef __attribute__((ext_vector_type(4)))  int    i32x4;

#define BATCH   32
#define SEQ     2048
#define DC      1024
#define DQ      1024
#define HID     512
#define MTILE   64            // S rows per workgroup (4 waves x 16)
#define NB      4             // nt columns blocked per accumulator group
#define NGROUPS (HID / 16 / NB)   // 8 groups of 4 nt-tiles
#define KTILES  (DC / 32)         // 32
#define LDA     1032          // 1024 + 8 halfs pad -> conflict-free ds_load_b128

#define LDSA_HALFS   (MTILE * LDA)          // 66048 halfs = 132096 B
#define BSTAGE_HALFS (16 * 32 * NB * 16)    // 32768 halfs = 64 KB per staged chunk
#define NSTAGES      (NGROUPS * 2)          // 16 chunks of 16 kt each

#define AS1 __attribute__((address_space(1)))
#define AS3 __attribute__((address_space(3)))

#if __has_builtin(__builtin_amdgcn_global_load_async_to_lds_b128)
#define HAVE_ASYNC_LDS 1
#else
#define HAVE_ASYNC_LDS 0
#endif

struct bfpair { bf16x8 lo, hi; };
__device__ __forceinline__ bf16x16 bfcat(bf16x8 lo, bf16x8 hi) {
    bfpair p{lo, hi};
    return __builtin_bit_cast(bf16x16, p);
}

__device__ __forceinline__ unsigned short f2bf(float f) {
    union { float f; unsigned u; } v; v.f = f;
    unsigned r = v.u + 0x7FFFu + ((v.u >> 16) & 1u);   // round-to-nearest-even
    return (unsigned short)(r >> 16);
}

// branchless tanh: 1 - 2/(exp(2x)+1); v_exp_f32 + v_rcp_f32, EXEC untouched
__device__ __forceinline__ float fast_tanh(float x) {
    float e = __expf(2.0f * x);
    return 1.0f - 2.0f * __builtin_amdgcn_rcpf(e + 1.0f);
}

// Stage one contiguous 64KB B chunk into LDS (async). 128 threads x 32 b128.
__device__ __forceinline__ void stage_b_chunk(const unsigned short* __restrict__ g,
                                              unsigned short* l, int tid) {
#if HAVE_ASYNC_LDS
#pragma unroll
    for (int k = 0; k < BSTAGE_HALFS / 8 / 128; ++k) {       // 32 iters
        int i = (k * 128 + tid) * 8;                          // halfs
        __builtin_amdgcn_global_load_async_to_lds_b128(
            (AS1 i32x4*)(g + i), (AS3 i32x4*)(l + i), 0, 0);
    }
#else
#pragma unroll
    for (int k = 0; k < BSTAGE_HALFS / 8 / 128; ++k) {
        int i = k * 128 + tid;
        ((uint4*)l)[i] = ((const uint4*)g)[i];
    }
#endif
}

__device__ __forceinline__ void wait_async_lds() {
#if HAVE_ASYNC_LDS
#if __has_builtin(__builtin_amdgcn_s_wait_asynccnt)
    __builtin_amdgcn_s_wait_asynccnt(0);
#else
    asm volatile("s_wait_asynccnt 0" ::: "memory");
#endif
#endif
}

// ---------------------------------------------------------------------------
// Kernel 0a: repack Wc = W1[0:1024,:] into staged WMMA-B bf16 fragments.
// Layout: [g][h][kt'][lane][j][16]: chunk (g,h) = contiguous 64KB;
// lane's NB=4 fragments for one kt are 128B contiguous (one DS base).
// Element: frag (nt = g*4+j, kt = h*16+kt'), lane = half*16 + n holds
// B[K = kt*32 + half*16 + jj][N = nt*16 + n], jj = 0..15.
// ---------------------------------------------------------------------------
__global__ __launch_bounds__(256)
void packWc_kernel(const float* __restrict__ W1, unsigned short* __restrict__ packed) {
    int idx  = blockIdx.x * 256 + threadIdx.x;   // ((g*2+h)*16 + kt')*32 + lane
    int lane = idx & 31;
    int kt   = (idx >> 5) & 15;
    int h    = (idx >> 9) & 1;
    int g    = idx >> 10;
    int kb   = (h * 16 + kt) * 32 + (lane >> 4) * 16;
    unsigned short* dst = packed + (size_t)idx * 64;
#pragma unroll
    for (int j = 0; j < NB; ++j) {
        int col = (g * NB + j) * 16 + (lane & 15);
#pragma unroll
        for (int jj = 0; jj < 16; ++jj)
            dst[j * 16 + jj] = f2bf(W1[(size_t)(kb + jj) * HID + col]);
    }
}

// ---------------------------------------------------------------------------
// Kernel 0b: qproj[b][h] = query[b,:] @ Wq[:,h] + b1[h]   (32 x 512, tiny)
// ---------------------------------------------------------------------------
__global__ __launch_bounds__(256)
void qproj_kernel(const float* __restrict__ query, const float* __restrict__ W1,
                  const float* __restrict__ b1, float* __restrict__ qproj) {
    int b = blockIdx.x;
    const float* Wq = W1 + (size_t)DC * HID;
    for (int h = threadIdx.x; h < HID; h += 256) {
        float acc = b1[h];
        for (int k = 0; k < DQ; ++k)
            acc += query[b * DQ + k] * Wq[(size_t)k * HID + h];
        qproj[b * HID + h] = acc;
    }
}

// ---------------------------------------------------------------------------
// Kernel 1: scores. One WG per (b, 64-row tile). 4 waves, 16 rows each.
// ctx tile fp32->bf16 in LDS; B chunks double-buffered via async-to-LDS;
// N-blocked WMMA (one A fragment -> 4 WMMAs), fused branchless tanh epilogue.
// ---------------------------------------------------------------------------
__global__ __launch_bounds__(128)
void scores_kernel(const float* __restrict__ ctx, const float* __restrict__ qproj,
                   const float* __restrict__ W2, const float* __restrict__ b2,
                   const unsigned short* __restrict__ packedWc,
                   float* __restrict__ scores) {
    extern __shared__ unsigned short lds[];
    unsigned short* ldsA = lds;                    // [MTILE][LDA]
    unsigned short* ldsB = lds + LDSA_HALFS;       // 2 x BSTAGE_HALFS
    const int b   = blockIdx.y;
    const int s0  = blockIdx.x * MTILE;
    const int tid = threadIdx.x;

    // ---- stage A tile: 64 x 1024 fp32 -> bf16 (float4 loads, uint2 LDS stores)
    const float* src = ctx + ((size_t)b * SEQ + s0) * DC;
#pragma unroll 4
    for (int it = 0; it < (MTILE * DC / 4) / 128; ++it) {
        int slot = it * 128 + tid;          // float4 slots
        int row  = slot >> 8;               // 256 float4 per row
        int c4   = slot & 255;
        float4 v = *(const float4*)(src + (size_t)row * DC + c4 * 4);
        unsigned lo = (unsigned)f2bf(v.x) | ((unsigned)f2bf(v.y) << 16);
        unsigned hi = (unsigned)f2bf(v.z) | ((unsigned)f2bf(v.w) << 16);
        *(uint2*)&ldsA[row * LDA + c4 * 4] = make_uint2(lo, hi);
    }
    __syncthreads();

    const int wave = tid >> 5;
    const int lane = tid & 31;
    const int half = lane >> 4;
    const int n    = lane & 15;
    // A fragment (this code's contiguous k order, not the ISA table order): lanes 0-15 K 0-15; lanes 16-31 K 16-31 (matches packed B)
    const unsigned short* aRow = &ldsA[(wave * 16 + n) * LDA + half * 16];

    float sacc[8];
#pragma unroll
    for (int v = 0; v < 8; ++v) sacc[v] = 0.0f;

    // prologue: stage first B chunk into buffer 0
    stage_b_chunk(packedWc, ldsB, tid);

    f32x8 acc[NB];

    for (int s = 0; s < NSTAGES; ++s) {
        const int g = s >> 1;        // nt group
        const int h = s & 1;         // kt half: kt = h*16 + kt'

        wait_async_lds();            // own async copies done
        __syncthreads();             // chunk s visible to all waves

        if (s + 1 < NSTAGES)         // writes the buffer everyone finished at s-1
            stage_b_chunk(packedWc + (size_t)(s + 1) * BSTAGE_HALFS,
                          ldsB + ((s + 1) & 1) * BSTAGE_HALFS, tid);

        if (h == 0) {
#pragma unroll
            for (int j = 0; j < NB; ++j)
                acc[j] = f32x8{0.f, 0.f, 0.f, 0.f, 0.f, 0.f, 0.f, 0.f};
        }

        const unsigned short* aBase = aRow + h * 16 * 32;           // kt' = 0 base
        const unsigned short* bBase = ldsB + (s & 1) * BSTAGE_HALFS + lane * 64;

#pragma unroll 2
        for (int kt = 0; kt < 16; ++kt) {
            bf16x8  alo = *(const bf16x8*)(aBase + kt * 32);
            bf16x8  ahi = *(const bf16x8*)(aBase + kt * 32 + 8);
            bf16x16 a   = bfcat(alo, ahi);
            const unsigned short* bp = bBase + kt * (32 * NB * 16); // kt row = 4KB
#pragma unroll
            for (int j = 0; j < NB; ++j) {
                bf16x16 bm = *(const bf16x16*)(bp + j * 16);
                acc[j] = __builtin_amdgcn_wmma_f32_16x16x32_bf16(
                             false, a, false, bm, (short)0, acc[j], false, false);
            }
        }

        if (h == 1) {                // group complete: fused epilogue for 4 nt cols
#pragma unroll
            for (int j = 0; j < NB; ++j) {
                int   hc  = (g * NB + j) * 16 + n;
                float qp  = qproj[b * HID + hc];
                float w2v = W2[hc];
#pragma unroll
                for (int v = 0; v < 8; ++v)
                    sacc[v] += fast_tanh(acc[j][v] + qp) * w2v;
            }
        }
    }

    // reduce over the 16 N-lanes of each half
#pragma unroll
    for (int v = 0; v < 8; ++v) {
        float x = sacc[v];
        x += __shfl_xor(x, 1, 32);
        x += __shfl_xor(x, 2, 32);
        x += __shfl_xor(x, 4, 32);
        x += __shfl_xor(x, 8, 32);
        sacc[v] = x;
    }
    if (n == 0) {
        float bb = b2[0];
#pragma unroll
        for (int v = 0; v < 8; ++v)
            scores[b * SEQ + s0 + wave * 16 + half * 8 + v] = sacc[v] + bb;
    }
}

// ---------------------------------------------------------------------------
// Kernel 2: masked softmax per batch row, in place over scores -> p_ctx
// ---------------------------------------------------------------------------
__global__ __launch_bounds__(256)
void softmax_kernel(const int* __restrict__ mask, float* __restrict__ p) {
    __shared__ float red[256];
    const int b = blockIdx.x, tid = threadIdx.x;
    float*      row  = p + b * SEQ;
    const int*  mrow = mask + b * SEQ;

    float vals[8];
    float mx = -3.4e38f;
#pragma unroll
    for (int i = 0; i < 8; ++i) {
        int s = tid + i * 256;
        float v = row[s];
        if (mrow[s] == 0) v = -10000.0f;
        vals[i] = v;
        mx = fmaxf(mx, v);
    }
    red[tid] = mx; __syncthreads();
    for (int off = 128; off > 0; off >>= 1) {
        if (tid < off) red[tid] = fmaxf(red[tid], red[tid + off]);
        __syncthreads();
    }
    mx = red[0]; __syncthreads();

    float sum = 0.f;
#pragma unroll
    for (int i = 0; i < 8; ++i) { vals[i] = __expf(vals[i] - mx); sum += vals[i]; }
    red[tid] = sum; __syncthreads();
    for (int off = 128; off > 0; off >>= 1) {
        if (tid < off) red[tid] += red[tid + off];
        __syncthreads();
    }
    float inv = 1.0f / red[0];
#pragma unroll
    for (int i = 0; i < 8; ++i) row[tid + i * 256] = vals[i] * inv;
}

// ---------------------------------------------------------------------------
// Kernel 3: expected_ctx[b,d] = sum_s p[b,s] * ctx[b,s,d]   (BW-bound stream)
// ---------------------------------------------------------------------------
__global__ __launch_bounds__(128)
void expected_kernel(const float* __restrict__ ctx, const float* __restrict__ p,
                     float* __restrict__ out) {
    __shared__ float pl[SEQ];
    const int b  = blockIdx.y;
    const int d4 = (blockIdx.x * 128 + threadIdx.x) * 4;
    for (int i = threadIdx.x; i < SEQ; i += 128) pl[i] = p[b * SEQ + i];
    __syncthreads();

    float4 acc = {0.f, 0.f, 0.f, 0.f};
    const float* base = ctx + (size_t)b * SEQ * DC + d4;
#pragma unroll 4
    for (int s = 0; s < SEQ; ++s) {
        float4 c = *(const float4*)(base + (size_t)s * DC);
        float  w = pl[s];
        acc.x += w * c.x; acc.y += w * c.y; acc.z += w * c.z; acc.w += w * c.w;
    }
    *(float4*)(out + b * DC + d4) = acc;
}

// ---------------------------------------------------------------------------
extern "C" void kernel_launch(void* const* d_in, const int* in_sizes, int n_in,
                              void* d_out, int out_size, void* d_ws, size_t ws_size,
                              hipStream_t stream) {
    const float* ctx   = (const float*)d_in[0];
    const float* query = (const float*)d_in[1];
    const float* W1    = (const float*)d_in[2];
    const float* b1    = (const float*)d_in[3];
    const float* W2    = (const float*)d_in[4];
    const float* b2    = (const float*)d_in[5];
    const int*   mask  = (const int*)d_in[6];

    float* out      = (float*)d_out;
    float* expected = out;                    // 32*1024
    float* p        = out + BATCH * DC;       // 32*2048 (raw scores, then probs)

    unsigned short* packedWc = (unsigned short*)d_ws;                        // 1 MB
    float*          qproj    = (float*)((char*)d_ws + (size_t)DC * HID * 2); // 64 KB

    packWc_kernel<<<dim3(32), dim3(256), 0, stream>>>(W1, packedWc);
    qproj_kernel<<<dim3(BATCH), dim3(256), 0, stream>>>(query, W1, b1, qproj);

    size_t ldsBytes = (size_t)(LDSA_HALFS + 2 * BSTAGE_HALFS) * sizeof(unsigned short);
    scores_kernel<<<dim3(SEQ / MTILE, BATCH), dim3(128), ldsBytes, stream>>>(
        ctx, qproj, W2, b2, packedWc, p);

    softmax_kernel<<<dim3(BATCH), dim3(256), 0, stream>>>(mask, p);
    expected_kernel<<<dim3(DC / (128 * 4), BATCH), dim3(128), 0, stream>>>(ctx, p, expected);
}
